// MultiHeadAttention_69441031242120
// MI455X (gfx1250) — matrix operand order repaired and checked by reading, not compiled, not run on hardware
//
#include <hip/hip_runtime.h>
#include <hip/hip_bf16.h>

// ---------------------------------------------------------------------------
// MHA forward for MI455X (gfx1250): bf16 WMMA + async global->LDS staging.
//   B=2, S=2048, D=1024, H=16, HD=64
// Pipeline:
//   [0] fp32 -> bf16 conversion of x, Wqkv, Wproj into workspace
//   [1] QKV GEMM  (4096x3072x1024) + bias  -> qkv bf16   (WMMA, dbl-buffered)
//       Q columns pre-scaled by HD^-0.5 * log2(e): softmax runs in base-2.
//   [2] flash attention per (b,h,128 q-rows)             (WMMA, dbl-buffered)
//       softmax denominator computed by the PV WMMA via a ones-column in V.
//   [3] proj GEMM (4096x1024x1024) + bias  -> out fp32   (WMMA, dbl-buffered)
// ---------------------------------------------------------------------------

typedef __bf16 bf16;
typedef __attribute__((ext_vector_type(16))) __bf16 bf16x16;
typedef __attribute__((ext_vector_type(8)))  __bf16 bf16x8;
typedef __attribute__((ext_vector_type(8)))  float  f32x8;

#define DEV __device__ __forceinline__

static constexpr int B_  = 2;
static constexpr int S_  = 2048;
static constexpr int D_  = 1024;
static constexpr int H_  = 16;
static constexpr int HD_ = 64;
static constexpr int TD_ = 3 * D_;   // 3072

// HD^-0.5 * log2(e): folded into Q via the GEMM-1 epilogue.
static constexpr float QSCALE = 0.125f * 1.44269504088896340736f;

// --- async global->LDS copy (16B per lane), tracked by ASYNCcnt -------------
DEV void async_ld_b128(void* lds, const void* gptr) {
  asm volatile("global_load_async_to_lds_b128 %0, %1, off"
               :: "v"((unsigned)(uintptr_t)lds),
                  "v"((unsigned long long)(uintptr_t)gptr)
               : "memory");
}

#if defined(__has_builtin) && __has_builtin(__builtin_amdgcn_s_wait_asynccnt)
#define WAIT_ASYNC() __builtin_amdgcn_s_wait_asynccnt(0)
#else
#define WAIT_ASYNC() asm volatile("s_wait_asynccnt 0x0" ::: "memory")
#endif

// --- WMMA wrapper: D = A(16x32 bf16) * B(32x16 bf16) + C(16x16 f32) ---------
DEV f32x8 wmma_bf16(bf16x16 a, bf16x16 b, f32x8 c) {
  return __builtin_amdgcn_wmma_f32_16x16x32_bf16(
      /*neg_a=*/false, a, /*neg_b=*/false, b,
      /*c_mod=*/(short)0, c, /*reuse_a=*/false, /*reuse_b=*/false);
}

// --- A-side fragment (16x32): lane = M row (lane&15);
//     per-lane K = {hf*8 .. hf*8+7} U {16+hf*8 .. 16+hf*8+7}  (ISA 7.12.2) ---
DEV bf16x16 frag_a(const bf16* base, int stride, int koff, int lane) {
  const int lo = lane & 15, hf = lane >> 4;
  const bf16* p = base + lo * stride + koff + hf * 8;
  union { bf16x16 v; bf16x8 h[2]; } u;
  u.h[0] = *(const bf16x8*)(p);
  u.h[1] = *(const bf16x8*)(p + 16);
  return u.v;
}

// --- B-side fragment (32x16): lane = N col (lane&15);
//     per-lane K = {hf*8 .. hf*8+7} U {16+hf*8 .. 16+hf*8+7} from K-major (B^T) storage ---
DEV bf16x16 frag_b(const bf16* base, int stride, int koff, int lane) {
  const int lo = lane & 15, hf = lane >> 4;
  const bf16* p = base + lo * stride + koff + hf * 8;
  union { bf16x16 v; bf16x8 h[2]; } u;
  u.h[0] = *(const bf16x8*)(p);
  u.h[1] = *(const bf16x8*)(p + 16);
  return u.v;
}

// ---------------------------------------------------------------------------
// [0] fp32 -> bf16
// ---------------------------------------------------------------------------
__global__ void cvt_f32_bf16(const float* __restrict__ in,
                             bf16* __restrict__ out, int n) {
  int i = blockIdx.x * blockDim.x + threadIdx.x;
  const int stride = gridDim.x * blockDim.x;
  for (; i < n; i += stride) out[i] = (bf16)in[i];
}

// ---------------------------------------------------------------------------
// [1]/[3] GEMM + bias (+ per-column scale for cols < scale_cols):
//   C[M,N] = (A[M,K] * Bm[K,N] + bias[N]) * (col < scale_cols ? scale : 1)
//   block tile 128x128x32, 256 threads = 8 waves (2x4), wave tile 64x32.
//   Double-buffered LDS; A staged with async-to-LDS, B^T staged via VGPRs.
// ---------------------------------------------------------------------------
template <typename OutT>
__global__ __launch_bounds__(256) void gemm_bias_wmma(
    const bf16* __restrict__ A, const bf16* __restrict__ Bm,
    const float* __restrict__ bias, OutT* __restrict__ C,
    int M, int N, int K, int scale_cols, float scale) {
  constexpr int BM = 128, BN = 128, BK = 32, LD = 40;  // LD: padded K stride
  __shared__ __align__(16) bf16 As[2][BM * LD];   // [m][k]
  __shared__ __align__(16) bf16 Bs[2][BN * LD];   // [n][k]  (B^T staging)

  const int tid = threadIdx.x, lane = tid & 31, wid = tid >> 5;
  const int wm0 = (wid >> 2) * 64;   // wave M origin in tile
  const int wn0 = (wid & 3) * 32;    // wave N origin in tile
  const int m0 = blockIdx.y * BM, n0 = blockIdx.x * BN;

  f32x8 acc[4][2] = {};

  const int arow = tid >> 1, aseg = (tid & 1) * 16;  // A: 128 rows x 2 segs
  const int brow = tid >> 3, bseg = (tid & 7) * 16;  // B: 32 rows  x 8 segs

  auto stage = [&](int k0, int buf) {
    const bf16* gA = A + (size_t)(m0 + arow) * K + k0 + aseg;
    async_ld_b128(&As[buf][arow * LD + aseg], gA);
    async_ld_b128(&As[buf][arow * LD + aseg + 8], gA + 8);
    const bf16* gB = Bm + (size_t)(k0 + brow) * N + n0 + bseg;
    union { bf16x8 v[2]; bf16 e[16]; } u;
    u.v[0] = *(const bf16x8*)gB;
    u.v[1] = *(const bf16x8*)(gB + 8);
#pragma unroll
    for (int i = 0; i < 16; ++i) Bs[buf][(bseg + i) * LD + brow] = u.e[i];
  };

  const int nT = K / BK;
  stage(0, 0);
  for (int t = 0; t < nT; ++t) {
    WAIT_ASYNC();          // this wave's async LDS writes are visible
    __syncthreads();       // everyone's tile-t staging is done
    if (t + 1 < nT) stage((t + 1) * BK, (t + 1) & 1);
    if (t + 2 < nT)        // global_prefetch of the slab after next
      __builtin_prefetch(Bm + (size_t)((t + 2) * BK + brow) * N + n0 + bseg, 0, 0);

    const bf16* as = As[t & 1];
    const bf16* bs = Bs[t & 1];
    bf16x16 af[4];
#pragma unroll
    for (int i = 0; i < 4; ++i)
      af[i] = frag_a(&as[(wm0 + i * 16) * LD], LD, 0, lane);
#pragma unroll
    for (int j = 0; j < 2; ++j) {
      bf16x16 bfg = frag_b(&bs[(wn0 + j * 16) * LD], LD, 0, lane);
#pragma unroll
      for (int i = 0; i < 4; ++i) acc[i][j] = wmma_bf16(af[i], bfg, acc[i][j]);
    }
  }

  // Epilogue: C/D layout -> lane = N (lane&15), VGPR r -> M = r + 8*hf
  const int lo = lane & 15, hf = lane >> 4;
#pragma unroll
  for (int j = 0; j < 2; ++j) {
    const int col = n0 + wn0 + j * 16 + lo;
    const float bv = bias[col];
    const float cs = (col < scale_cols) ? scale : 1.0f;
#pragma unroll
    for (int i = 0; i < 4; ++i) {
#pragma unroll
      for (int r = 0; r < 8; ++r) {
        const int row = m0 + wm0 + i * 16 + r + 8 * hf;
        C[(size_t)row * N + col] = (OutT)((acc[i][j][r] + bv) * cs);
      }
    }
  }
}

// ---------------------------------------------------------------------------
// [2] flash attention. One block = (b, h, 128 query rows); 8 waves x 16 rows.
//     qkv layout: [b, s, 3, H, HD] (row stride TD_), out: [b, s, H*HD]
//     Q already carries HD^-0.5 * log2(e): softmax runs in base-2 (exp2f).
//     K tiles staged async-to-LDS, V^T via VGPR transpose; double-buffered.
//     V^T carries a 5th 16-row block: row 64 = 1.0, rows 65..79 = 0, so the
//     PV WMMA also produces the softmax denominator (O' = P @ [V | e]); the
//     running alpha-rescale maintains l = O'[:,64] across tiles for free.
// ---------------------------------------------------------------------------
__global__ __launch_bounds__(256) void flash_attn_wmma(
    const bf16* __restrict__ qkv, bf16* __restrict__ out) {
  constexpr int KT = 64;   // key tile
  constexpr int LK = 72;   // padded stride (144B, 16B-aligned runs)
  constexpr int VR = HD_ + 16;   // V^T rows incl. ones-block

  __shared__ __align__(16) bf16 Ks[2][KT * LK];     // [key][hd]
  __shared__ __align__(16) bf16 Vt[2][VR * LK];     // [hd | ones][key]
  __shared__ __align__(16) bf16 Ps[8 * 16 * LK];    // per-wave P strips

  const int tid = threadIdx.x, lane = tid & 31, wid = tid >> 5;
  const int lo = lane & 15, hf = lane >> 4;
  const int b = blockIdx.z, h = blockIdx.y;
  const int q0 = blockIdx.x * 128 + wid * 16;

  // One-time fill of the ones-block (rows HD_..HD_+15 of both V^T buffers).
  for (int idx = tid; idx < 16 * KT; idx += 256) {
    const int i = idx >> 6, key = idx & (KT - 1);
    const bf16 v = (i == 0) ? (bf16)1.0f : (bf16)0.0f;
    Vt[0][(HD_ + i) * LK + key] = v;
    Vt[1][(HD_ + i) * LK + key] = v;
  }

  // Q fragments: rows contiguous in memory (stride TD_), kept in registers.
  const bf16* qbase = qkv + ((size_t)b * S_ + q0) * TD_ + h * HD_;
  bf16x16 qf[2];
  qf[0] = frag_a(qbase, TD_, 0, lane);
  qf[1] = frag_a(qbase, TD_, 32, lane);

  f32x8 o[5] = {};            // o[0..3] = output blocks, o[4] = denominator
  float mrun[8];
#pragma unroll
  for (int r = 0; r < 8; ++r) mrun[r] = -1e30f;

  const int krow = tid >> 2, kseg = (tid & 3) * 16;  // staging map: 64 x 4

  auto stage = [&](int kt, int buf) {
    const bf16* gK = qkv + ((size_t)b * S_ + kt + krow) * TD_ + D_ + h * HD_ + kseg;
    async_ld_b128(&Ks[buf][krow * LK + kseg], gK);
    async_ld_b128(&Ks[buf][krow * LK + kseg + 8], gK + 8);
    const bf16* gV = qkv + ((size_t)b * S_ + kt + krow) * TD_ + 2 * D_ + h * HD_ + kseg;
    union { bf16x8 v[2]; bf16 e[16]; } u;
    u.v[0] = *(const bf16x8*)gV;
    u.v[1] = *(const bf16x8*)(gV + 8);
#pragma unroll
    for (int i = 0; i < 16; ++i) Vt[buf][(kseg + i) * LK + krow] = u.e[i];
  };

  const int nIt = S_ / KT;
  stage(0, 0);
  for (int it = 0; it < nIt; ++it) {
    WAIT_ASYNC();
    __syncthreads();
    if (it + 1 < nIt) stage((it + 1) * KT, (it + 1) & 1);
    if (it + 2 < nIt)
      __builtin_prefetch(
          qkv + ((size_t)b * S_ + (it + 2) * KT + krow) * TD_ + 2 * D_ + h * HD_ + kseg,
          0, 0);

    const bf16* ks = Ks[it & 1];
    const bf16* vt = Vt[it & 1];

    // scores S = Q (16xHD) @ K^T (HDx64), already in log2 units.
    f32x8 sc[4] = {};
#pragma unroll
    for (int j = 0; j < 4; ++j) {
#pragma unroll
      for (int kc = 0; kc < 2; ++kc) {
        bf16x16 kf = frag_b(&ks[j * 16 * LK], LK, kc * 32, lane);
        sc[j] = wmma_bf16(qf[kc], kf, sc[j]);
      }
    }

    // base-2 online softmax (max reduction only; sums come from the WMMA).
    bf16* pl = &Ps[wid * 16 * LK];
#pragma unroll
    for (int r = 0; r < 8; ++r) {
      const float s0 = sc[0][r], s1 = sc[1][r], s2 = sc[2][r], s3 = sc[3][r];
      float mr = fmaxf(fmaxf(s0, s1), fmaxf(s2, s3));
#pragma unroll
      for (int off = 1; off < 16; off <<= 1)
        mr = fmaxf(mr, __shfl_xor(mr, off, 32));
      const float mnew  = fmaxf(mrun[r], mr);
      const float alpha = exp2f(mrun[r] - mnew);
      mrun[r] = mnew;
#pragma unroll
      for (int jn = 0; jn < 5; ++jn) o[jn][r] *= alpha;   // incl. denominator
      const int m = r + 8 * hf;   // C-layout row -> P strip row
      pl[m * LK +  0 + lo] = (bf16)exp2f(s0 - mnew);
      pl[m * LK + 16 + lo] = (bf16)exp2f(s1 - mnew);
      pl[m * LK + 32 + lo] = (bf16)exp2f(s2 - mnew);
      pl[m * LK + 48 + lo] = (bf16)exp2f(s3 - mnew);
    }

    // O' += P (16x64) @ [V | e] (64 x HD+16): block jn=4 accumulates l.
#pragma unroll
    for (int kc = 0; kc < 2; ++kc) {
      bf16x16 pf = frag_a(pl, LK, kc * 32, lane);
#pragma unroll
      for (int jn = 0; jn < 5; ++jn) {
        bf16x16 vf = frag_b(&vt[jn * 16 * LK], LK, kc * 32, lane);
        o[jn] = wmma_bf16(pf, vf, o[jn]);
      }
    }
  }

  // normalize and store: out[b, s, h*HD + hd].
  // l for row (r, hf) sits in o[4][r] at column 0 of block 4 = lane hf*16.
#pragma unroll
  for (int r = 0; r < 8; ++r) {
    const float l   = __shfl(o[4][r], lane & 16, 32);
    const float inv = 1.f / l;
    const int s = q0 + r + 8 * hf;
#pragma unroll
    for (int jn = 0; jn < 4; ++jn)
      out[((size_t)b * S_ + s) * D_ + h * HD_ + jn * 16 + lo] =
          (bf16)(o[jn][r] * inv);
  }
}

// ---------------------------------------------------------------------------
// Launcher
// ---------------------------------------------------------------------------
extern "C" void kernel_launch(void* const* d_in, const int* in_sizes, int n_in,
                              void* d_out, int out_size, void* d_ws,
                              size_t ws_size, hipStream_t stream) {
  (void)in_sizes; (void)n_in; (void)out_size; (void)ws_size;
  const float* x     = (const float*)d_in[0];   // [B,S,D]
  const float* Wqkv  = (const float*)d_in[1];   // [D,3D]
  const float* bqkv  = (const float*)d_in[2];   // [3D]
  const float* Wproj = (const float*)d_in[3];   // [D,D]
  const float* bproj = (const float*)d_in[4];   // [D]

  const size_t nX = (size_t)B_ * S_ * D_;       // 4,194,304
  const size_t nWq = (size_t)D_ * TD_;          // 3,145,728
  const size_t nWp = (size_t)D_ * D_;           // 1,048,576
  const size_t nQkv = (size_t)B_ * S_ * TD_;    // 12,582,912

  char* ws = (char*)d_ws;                       // ~48 MB total
  bf16* x_bf    = (bf16*)(ws);
  bf16* wqkv_bf = (bf16*)(ws + 2 * nX);
  bf16* wproj_bf= (bf16*)(ws + 2 * (nX + nWq));
  bf16* qkv_bf  = (bf16*)(ws + 2 * (nX + nWq + nWp));
  bf16* attn_bf = (bf16*)(ws + 2 * (nX + nWq + nWp + nQkv));

  cvt_f32_bf16<<<1024, 256, 0, stream>>>(x, x_bf, (int)nX);
  cvt_f32_bf16<<<1024, 256, 0, stream>>>(Wqkv, wqkv_bf, (int)nWq);
  cvt_f32_bf16<<<512, 256, 0, stream>>>(Wproj, wproj_bf, (int)nWp);

  // [1] qkv = x @ Wqkv + bqkv, Q cols scaled by HD^-0.5*log2(e)
  gemm_bias_wmma<bf16><<<dim3(TD_ / 128, (B_ * S_) / 128), 256, 0, stream>>>(
      x_bf, wqkv_bf, bqkv, qkv_bf, B_ * S_, TD_, D_, D_, QSCALE);

  // [2] attention
  flash_attn_wmma<<<dim3(S_ / 128, H_, B_), 256, 0, stream>>>(qkv_bf, attn_bf);

  // [3] out = attn @ Wproj + bproj    (M=4096, N=1024, K=1024), fp32 out
  gemm_bias_wmma<float><<<dim3(D_ / 128, (B_ * S_) / 128), 256, 0, stream>>>(
      attn_bf, wproj_bf, bproj, (float*)d_out, B_ * S_, D_, D_, 0, 1.0f);
}
